// QuantizedLinear_12086037971014
// MI455X (gfx1250) — matrix operand order repaired and checked by reading, not compiled, not run on hardware
//
#include <hip/hip_runtime.h>
#include <hip/hip_bf16.h>

// GEMM: M=8192 (B*S), N=4096 (OUT), K=4096 (IN)
// y = x @ (scale*(wq - zp))^T + bias, via V_WMMA_F32_16X16X32_F16.
// Block tile 128(M) x 256(N), 8 wave32s in 2x4, wave tile 64x64 = 16 WMMA/K-step.

typedef __attribute__((ext_vector_type(16))) _Float16 v16h;
typedef __attribute__((ext_vector_type(8)))  _Float16 v8h;
typedef __attribute__((ext_vector_type(8)))  float    v8f;
typedef __attribute__((ext_vector_type(4)))  float    f32x4;
typedef __attribute__((ext_vector_type(4)))  int      i32x4;
typedef __attribute__((ext_vector_type(4)))  _Float16 h16x4;

#define BM 128
#define BN 256
#define BK 32
#define LDSS 40   // LDS row stride in halves (BK + 8 pad to spread banks)

__global__ __launch_bounds__(256)
void QuantizedLinear_wmma_kernel(const float* __restrict__ x,
                                 const int*   __restrict__ wq,
                                 const float* __restrict__ bias,
                                 const float* __restrict__ scale_p,
                                 const float* __restrict__ zp_p,
                                 float* __restrict__ out,
                                 int M, int N, int K)
{
    __shared__ _Float16 As[2][BM * LDSS];   // x tile, f16, [m][k]
    __shared__ _Float16 Bs[2][BN * LDSS];   // dequantized w tile, f16, [n][k]

    const int tid      = threadIdx.x;
    const int lane     = tid & 31;          // wave32
    const int wave     = tid >> 5;          // 0..7
    const int waveM    = wave & 1;          // 0..1 -> 64 rows each
    const int waveN    = wave >> 1;         // 0..3 -> 64 cols each
    const int half_sel = lane >> 4;         // lane group 0-15 / 16-31
    const int l16      = lane & 15;

    const int blockN = blockIdx.x * BN;
    const int blockM = blockIdx.y * BM;

    const float scale = scale_p[0];
    const float nszp  = -scale * zp_p[0];   // w_fp = fma(q, scale, nszp)

    // A tile: 128x32 f32  = 1024 vec4 -> 4 per thread
    // B tile: 256x32 i32  = 2048 vec4 -> 8 per thread
    f32x4 aReg[4];
    i32x4 bReg[8];

    auto loadTiles = [&](int kBase) {
        #pragma unroll
        for (int i = 0; i < 4; ++i) {
            const int id  = tid + i * 256;
            const int row = id >> 3;              // 0..127
            const int cv  = (id & 7) * 4;         // 0,4,...,28
            aReg[i] = *(const f32x4*)(x + (size_t)(blockM + row) * K + kBase + cv);
        }
        #pragma unroll
        for (int i = 0; i < 8; ++i) {
            const int id  = tid + i * 256;
            const int row = id >> 3;              // 0..255
            const int cv  = (id & 7) * 4;
            bReg[i] = *(const i32x4*)(wq + (size_t)(blockN + row) * K + kBase + cv);
        }
    };
    auto stashTiles = [&](int buf) {
        #pragma unroll
        for (int i = 0; i < 4; ++i) {
            const int id  = tid + i * 256;
            const int row = id >> 3;
            const int cv  = (id & 7) * 4;
            h16x4 ah;
            #pragma unroll
            for (int j = 0; j < 4; ++j) ah[j] = (_Float16)aReg[i][j];
            *(h16x4*)(&As[buf][row * LDSS + cv]) = ah;
        }
        #pragma unroll
        for (int i = 0; i < 8; ++i) {
            const int id  = tid + i * 256;
            const int row = id >> 3;
            const int cv  = (id & 7) * 4;
            h16x4 bh;
            #pragma unroll
            for (int j = 0; j < 4; ++j)
                bh[j] = (_Float16)__builtin_fmaf((float)bReg[i][j], scale, nszp);
            *(h16x4*)(&Bs[buf][row * LDSS + cv]) = bh;
        }
    };

    v8f acc[4][4] = {};   // 4 (M tiles) x 4 (N tiles) of 16x16 f32 = 128 VGPRs

    const int NK = K / BK;

    loadTiles(0);
    stashTiles(0);
    __syncthreads();

    union F16frag { v16h v; struct { v8h lo, hi; } p; };

    for (int kb = 0; kb < NK; ++kb) {
        const int cur = kb & 1;

        // Prefetch next global tiles while this buffer feeds the WMMAs.
        if (kb + 1 < NK) loadTiles((kb + 1) * BK);

        // B fragments (B 32x16 layout): lanes 0-15 hold K 0..7 / 16..23,
        // lanes 16-31 hold K 8..15 / 24..31, column N = lane%16.
        F16frag bfrag[4];
        #pragma unroll
        for (int tn = 0; tn < 4; ++tn) {
            const _Float16* pb =
                &Bs[cur][(waveN * 64 + tn * 16 + l16) * LDSS + 8 * half_sel];
            bfrag[tn].p.lo = *(const v8h*)(pb);
            bfrag[tn].p.hi = *(const v8h*)(pb + 16);
        }

        #pragma unroll
        for (int tm = 0; tm < 4; ++tm) {
            // A fragment (A 16x32 layout): lanes 0-15 hold K 0..7 / 16..23,
            // lanes 16-31 hold K 8..15 / 24..31, row M = lane%16.
            F16frag afrag;
            const _Float16* pa =
                &As[cur][(waveM * 64 + tm * 16 + l16) * LDSS + 8 * half_sel];
            afrag.p.lo = *(const v8h*)(pa);        // K = 8*hs .. +7
            afrag.p.hi = *(const v8h*)(pa + 16);   // K = 16+8*hs .. +7

            #pragma unroll
            for (int tn = 0; tn < 4; ++tn)
                acc[tm][tn] = __builtin_amdgcn_wmma_f32_16x16x32_f16(
                    /*neg_a=*/false, afrag.v,
                    /*neg_b=*/false, bfrag[tn].v,
                    /*c_mod=*/(short)0, acc[tm][tn],
                    /*reuse_a=*/false, /*reuse_b=*/false);
        }

        if (kb + 1 < NK) stashTiles((kb + 1) & 1);
        __syncthreads();
    }

    // Epilogue: C/D layout -> VGPR r holds M = r + 8*half_sel, N = lane%16.
    #pragma unroll
    for (int tm = 0; tm < 4; ++tm) {
        const int mBase = blockM + waveM * 64 + tm * 16 + 8 * half_sel;
        #pragma unroll
        for (int tn = 0; tn < 4; ++tn) {
            const int n  = blockN + waveN * 64 + tn * 16 + l16;
            const float bv = bias[n];
            #pragma unroll
            for (int r = 0; r < 8; ++r) {
                out[(size_t)(mBase + r) * N + n] = acc[tm][tn][r] + bv;
            }
        }
    }
}

extern "C" void kernel_launch(void* const* d_in, const int* in_sizes, int n_in,
                              void* d_out, int out_size, void* d_ws, size_t ws_size,
                              hipStream_t stream) {
    const float* x    = (const float*)d_in[0];   // [B, S, IN] f32
    const int*   wq   = (const int*)  d_in[1];   // [OUT, IN] int32
    const float* bias = (const float*)d_in[2];   // [OUT] f32
    const float* sc   = (const float*)d_in[3];   // scalar
    const float* zp   = (const float*)d_in[4];   // scalar
    float* out = (float*)d_out;                  // [B, S, OUT] f32

    const int K = 4096;                          // IN
    const int N = in_sizes[2];                   // OUT = 4096
    const int M = in_sizes[0] / K;               // B*S = 8192

    dim3 grid(N / BN, M / BM);                   // (16, 64)
    QuantizedLinear_wmma_kernel<<<grid, dim3(256), 0, stream>>>(
        x, wq, bias, sc, zp, out, M, N, K);
}
